// MAMFE_Block_20873541058723
// MI455X (gfx1250) — compile-verified, matrix operand order checked by reading, not run on hardware
//
#include <hip/hip_runtime.h>

typedef unsigned short u16;
typedef unsigned int   u32;
typedef __attribute__((ext_vector_type(16))) __bf16 v16b;
typedef __attribute__((ext_vector_type(8)))  float  v8f;
typedef __attribute__((ext_vector_type(8)))  u16    v8u;
typedef __attribute__((ext_vector_type(4)))  int    v4i;

#define DIMF 512
#define D2F  256
#define NHH  4
#define DHH  32
#define SEGL 64
#define AWL  68
#define AWP  80        // rows padded to multiple of 16
#define KP3  96        // K=68 padded to multiple of 32
#define NSEGS 256
#define NTOK 17408     // tokens per batch per stream (NSEGS*AWL)
#define NTS  34816     // tokens per stream (2 batches)
#define HPIX 128
#define WPIX 128
#define NWIN 2
#define NPIX 16384
#define TOKY 32768

// ---------- bf16 helpers (round-to-nearest-even) ----------
__device__ __forceinline__ u16 f2b(float f){
  u32 u = __builtin_bit_cast(u32, f);
  u32 r = u + 0x7FFFu + ((u >> 16) & 1u);
  return (u16)(r >> 16);
}
__device__ __forceinline__ float b2f(u16 h){
  u32 u = ((u32)h) << 16;
  return __builtin_bit_cast(float, u);
}

// ---------- async global->LDS copy (gfx1250 path when available) ----------
#if __has_builtin(__builtin_amdgcn_global_load_async_to_lds_b128)
#define USE_ASYNC_CP 1
typedef __attribute__((address_space(1))) v4i* gp128;
typedef __attribute__((address_space(3))) v4i* lp128;
#endif

__device__ __forceinline__ void cp16(void* l, const void* g){
#ifdef USE_ASYNC_CP
  __builtin_amdgcn_global_load_async_to_lds_b128((gp128)g, (lp128)l, 0, 0);
#else
  *(v8u*)l = *(const v8u*)g;
#endif
}
__device__ __forceinline__ void cp_wait(){
#ifdef USE_ASYNC_CP
#if __has_builtin(__builtin_amdgcn_s_wait_asynccnt)
  __builtin_amdgcn_s_wait_asynccnt(0);
#else
  asm volatile("s_wait_asynccnt 0x0" ::: "memory");
#endif
#endif
}

// ---------- WMMA fragment helpers (CDNA5 wave32 16x16x32 bf16) ----------
// A (16x32): m = lane&15 (both halves), per-lane K elements:
//   e<8 -> K = g*8+e ; e>=8 -> K = 16+g*8+(e-8)   (g = lane>>4)
// B (32x16): mirrored; lane covers n = lane&15, elements along K.
// -> with a K-contiguous line, a fragment is exactly two 16B ds loads.
__device__ __forceinline__ v16b frag_ld(const u16* p, int kbase, int g){
  union { v8u h[2]; v16b v; } r;
  const u16* q = p + kbase + g * 8;
  r.h[0] = *(const v8u*)(q);
  r.h[1] = *(const v8u*)(q + 16);
  return r.v;
}
__device__ __forceinline__ v8f wmma_bf16(v16b a, v16b b, v8f c){
  return __builtin_amdgcn_wmma_f32_16x16x32_bf16(false, a, false, b, (short)0, c,
                                                 false, false);
}

// =====================================================================
// 1) Build x0m/x1m (+mems +pos_emb); write f32 accumulator and bf16 copy
// =====================================================================
__global__ void __launch_bounds__(256)
k_build_xm(const float* __restrict__ x, const float* __restrict__ mem0,
           const float* __restrict__ mem1, const float* __restrict__ pe,
           float* __restrict__ xo, u16* __restrict__ xmb, int stream1){
  int t = blockIdx.x;
  int c = threadIdx.x;
  int b   = t / NTOK;
  int r   = t - b * NTOK;
  int seg = r / AWL;
  int pos = r - seg * AWL;
  float v;
  if (pos < SEGL){
    int a0 = seg / NWIN;
    int a1 = seg - a0 * NWIN;
    int hh, ww, cc;
    if (!stream1){ hh = a0; ww = a1 * SEGL + pos; cc = c; }
    else         { ww = a0; hh = a1 * SEGL + pos; cc = c + D2F; }
    v = x[(((long)b * HPIX + hh) * WPIX + ww) * DIMF + cc];
  } else {
    const float* m = (b == 0) ? mem0 : mem1;
    v = m[(pos - SEGL) * D2F + c];
  }
  v += pe[(long)r * D2F + c];
  long o = (long)t * D2F + c;
  xo[o]  = v;
  xmb[o] = f2b(v);
}

// =====================================================================
// 2) Windowed attention: one workgroup (8 waves) per 68-token window
// =====================================================================
__global__ void __launch_bounds__(256)
k_wattn(const u16* __restrict__ xmb, float* __restrict__ xo,
        const float* __restrict__ Wq, const float* __restrict__ Wk,
        const float* __restrict__ Wv, const float* __restrict__ Wo){
  __shared__ __align__(16) u16  Xs[AWP * D2F];    // 80x256, rows 68.. zero
  __shared__ __align__(16) u16  Wt[128 * 256];    // W^T staging (reused)
  __shared__ __align__(16) u16  Qb[AWP * 128];
  __shared__ __align__(16) u16  Kb[AWP * 128];    // row-major K == B^T
  __shared__ __align__(16) u16  Vt[128 * KP3];    // V^T  [feat][token]
  __shared__              float Sf[AWP * AWP];    // scores f32
  __shared__ __align__(16) u16  Pb[AWP * KP3];    // probs  [row][token]
  __shared__ __align__(16) u16  Ob[AWP * 128];

  int tid = threadIdx.x;
  int wv = tid >> 5, lane = tid & 31, g = lane >> 4, ln = lane & 15;
  long base = (long)blockIdx.x * AWL;

  { // stage X (async 16B chunks), zero pads
    const u16* src = xmb + base * D2F;
    for (int i = tid; i < AWL * D2F / 8; i += 256)
      cp16(Xs + i * 8, src + i * 8);
    for (int i = AWL * D2F + tid; i < AWP * D2F; i += 256) Xs[i] = 0;
    for (int i = tid; i < 128 * KP3; i += 256) Vt[i] = 0;
    for (int i = tid; i < AWP * KP3; i += 256) Pb[i] = 0;
    cp_wait();
  }
  __syncthreads();

  const float scale = 0.17677669529663687f; // 1/sqrt(32)

  // ---- Q/K/V projections ----
  for (int p = 0; p < 3; ++p){
    const float* W = (p == 0) ? Wq : (p == 1) ? Wk : Wv;
    for (int i = tid; i < 128 * 256; i += 256)
      Wt[i] = f2b(W[(i & 255) * 128 + (i >> 8)]);          // Wt[n][k]
    __syncthreads();
    for (int t = wv; t < 5 * 8; t += 8){
      int mt = t / 8, nt = t - mt * 8;
      int m = mt * 16 + ln, n = nt * 16 + ln;
      v8f c = {};
      #pragma unroll
      for (int kb = 0; kb < 256; kb += 32){
        v16b a = frag_ld(Xs + m * D2F, kb, g);
        v16b b = frag_ld(Wt + n * 256, kb, g);
        c = wmma_bf16(a, b, c);
      }
      #pragma unroll
      for (int rr = 0; rr < 8; ++rr){
        int row = mt * 16 + rr + 8 * g, col = nt * 16 + ln;
        float val = c[rr];                   // pad rows compute to 0
        if      (p == 0) Qb[row * 128 + col] = f2b(val * scale);
        else if (p == 1) Kb[row * 128 + col] = f2b(val);
        else             Vt[col * KP3 + row] = f2b(val);
      }
    }
    __syncthreads();
  }

  // ---- per-head attention ----
  for (int h = 0; h < NHH; ++h){
    // S = Qh @ Kh^T (single K-step of 32)
    for (int t = wv; t < 25; t += 8){
      int mt = t / 5, nt = t - mt * 5;
      int m = mt * 16 + ln, n = nt * 16 + ln;
      v8f c = {};
      v16b a = frag_ld(Qb + m * 128 + h * DHH, 0, g);
      v16b b = frag_ld(Kb + n * 128 + h * DHH, 0, g);
      c = wmma_bf16(a, b, c);
      #pragma unroll
      for (int rr = 0; rr < 8; ++rr){
        int row = mt * 16 + rr + 8 * g, col = nt * 16 + ln;
        Sf[row * AWP + col] = c[rr];
      }
    }
    __syncthreads();
    // softmax rows (k < 68 only; pads of Pb stay zero)
    for (int row = tid; row < AWL; row += 256){
      float mx = -1e30f;
      for (int j = 0; j < AWL; ++j) mx = fmaxf(mx, Sf[row * AWP + j]);
      float s = 0.f;
      for (int j = 0; j < AWL; ++j){
        float e = __expf(Sf[row * AWP + j] - mx);
        Sf[row * AWP + j] = e; s += e;
      }
      float inv = 1.f / s;
      for (int j = 0; j < AWL; ++j) Pb[row * KP3 + j] = f2b(Sf[row * AWP + j] * inv);
    }
    __syncthreads();
    // O_h = P @ V_h  (K padded to 96)
    for (int t = wv; t < 10; t += 8){
      int mt = t >> 1, nt = t & 1;
      int m = mt * 16 + ln;
      int ncol = h * DHH + nt * 16 + ln;
      v8f c = {};
      #pragma unroll
      for (int kb = 0; kb < KP3; kb += 32){
        v16b a = frag_ld(Pb + m * KP3, kb, g);
        v16b b = frag_ld(Vt + ncol * KP3, kb, g);
        c = wmma_bf16(a, b, c);
      }
      #pragma unroll
      for (int rr = 0; rr < 8; ++rr){
        int row = mt * 16 + rr + 8 * g, col = h * DHH + nt * 16 + ln;
        Ob[row * 128 + col] = f2b(c[rr]);
      }
    }
    __syncthreads();
  }

  // ---- out = O @ Wo, accumulate into xo ----
  for (int i = tid; i < 256 * 128; i += 256)
    Wt[i] = f2b(Wo[(i & 127) * 256 + (i >> 7)]);           // WoT[n][k]
  __syncthreads();
  for (int t = wv; t < 5 * 16; t += 8){
    int mt = t >> 4, nt = t & 15;
    int m = mt * 16 + ln, n = nt * 16 + ln;
    v8f c = {};
    #pragma unroll
    for (int kb = 0; kb < 128; kb += 32){
      v16b a = frag_ld(Ob + m * 128, kb, g);
      v16b b = frag_ld(Wt + n * 128, kb, g);
      c = wmma_bf16(a, b, c);
    }
    #pragma unroll
    for (int rr = 0; rr < 8; ++rr){
      int row = mt * 16 + rr + 8 * g, col = nt * 16 + ln;
      if (row < AWL){
        long o = (base + row) * D2F + col;
        xo[o] += c[rr];
      }
    }
  }
}

// =====================================================================
// 3) 256x256 projection (nmem Q/K/V), 64-token tiles
// =====================================================================
__global__ void __launch_bounds__(256)
k_proj256(const u16* __restrict__ A, const float* __restrict__ W,
          u16* __restrict__ out){
  __shared__ __align__(16) u16 As[64 * 256];
  __shared__ __align__(16) u16 Wt[256 * 256];
  int tid = threadIdx.x;
  int wv = tid >> 5, lane = tid & 31, g = lane >> 4, ln = lane & 15;
  long rbase = (long)blockIdx.x * 64;
  {
    const u16* src = A + rbase * 256;
    for (int i = tid; i < 64 * 256 / 8; i += 256)
      cp16(As + i * 8, src + i * 8);
  }
  for (int i = tid; i < 256 * 256; i += 256) Wt[i] = f2b(W[(i & 255) * 256 + (i >> 8)]);
  cp_wait();
  __syncthreads();
  for (int t = wv; t < 64; t += 8){
    int mt = t >> 4, nt = t & 15;
    int m = mt * 16 + ln, n = nt * 16 + ln;
    v8f c = {};
    #pragma unroll
    for (int kb = 0; kb < 256; kb += 32){
      v16b a = frag_ld(As + m * 256, kb, g);
      v16b b = frag_ld(Wt + n * 256, kb, g);
      c = wmma_bf16(a, b, c);
    }
    #pragma unroll
    for (int rr = 0; rr < 8; ++rr){
      int row = mt * 16 + rr + 8 * g, col = nt * 16 + ln;
      out[(rbase + row) * 256 + col] = f2b(c[rr]);
    }
  }
}

// =====================================================================
// 4) Neural-memory scan: M^T resident in LDS (bf16), K-contiguous
//    copies of K^T and E for the grad GEMM. 16 waves per batch.
// =====================================================================
__global__ void __launch_bounds__(512)
k_nmem(const u16* __restrict__ Qp, const u16* __restrict__ Kp,
       const u16* __restrict__ Vp, float* __restrict__ xo){
  __shared__ __align__(16) u16 Mt[256 * 256];   // Mt[e][d] = M[d][e]
  __shared__ __align__(16) u16 Qc[AWP * 256];   // [token][feat]
  __shared__ __align__(16) u16 Kc[AWP * 256];
  __shared__ __align__(16) u16 Kct[256 * KP3];  // [feat][token]
  __shared__ __align__(16) u16 Et[256 * KP3];   // [feat][token]
  int tid = threadIdx.x;
  int wv = tid >> 5, lane = tid & 31, g = lane >> 4, ln = lane & 15;
  long tb0 = (long)blockIdx.x * NTOK;

  for (int i = tid; i < 256 * 256; i += 512) Mt[i] = 0;
  for (int i = AWL * 256 + tid; i < AWP * 256; i += 512){ Qc[i] = 0; Kc[i] = 0; }
  for (int i = tid; i < 256 * KP3; i += 512){ Kct[i] = 0; Et[i] = 0; }
  __syncthreads();

  const float upd = -0.1f / (float)AWL;   // -LR/C
  for (int ch = 0; ch < NSEGS; ++ch){
    long rb = (tb0 + (long)ch * AWL) * 256;
    { // stage chunk: Q/K async; K also transposed (K-contiguous)
      const u16* qs = Qp + rb;
      const u16* ks = Kp + rb;
      for (int i = tid; i < AWL * 256 / 8; i += 512){
        cp16(Qc + i * 8, qs + i * 8);
        cp16(Kc + i * 8, ks + i * 8);
      }
      for (int i = tid; i < AWL * 256; i += 512){
        int c = i >> 8, d = i & 255;
        Kct[d * KP3 + c] = Kp[rb + i];
      }
      cp_wait();
    }
    __syncthreads();
    // ret = Q @ M  -> xo +=        (tiles 5x16)
    for (int t = wv; t < 80; t += 16){
      int mt = t >> 4, nt = t & 15;
      int m = mt * 16 + ln, n = nt * 16 + ln;
      v8f c = {};
      #pragma unroll
      for (int kb = 0; kb < 256; kb += 32){
        v16b a = frag_ld(Qc + m * 256, kb, g);
        v16b b = frag_ld(Mt + n * 256, kb, g);   // B[k=d][n=e] = Mt[e][d]
        c = wmma_bf16(a, b, c);
      }
      #pragma unroll
      for (int rr = 0; rr < 8; ++rr){
        int row = mt * 16 + rr + 8 * g, col = nt * 16 + ln;
        if (row < AWL) xo[(tb0 + (long)ch * AWL + row) * 256 + col] += c[rr];
      }
    }
    // E = K @ M - V   (stored transposed, K-contiguous)
    for (int t = wv; t < 80; t += 16){
      int mt = t >> 4, nt = t & 15;
      int m = mt * 16 + ln, n = nt * 16 + ln;
      v8f c = {};
      #pragma unroll
      for (int kb = 0; kb < 256; kb += 32){
        v16b a = frag_ld(Kc + m * 256, kb, g);
        v16b b = frag_ld(Mt + n * 256, kb, g);
        c = wmma_bf16(a, b, c);
      }
      #pragma unroll
      for (int rr = 0; rr < 8; ++rr){
        int row = mt * 16 + rr + 8 * g, col = nt * 16 + ln;
        if (row < AWL){
          float v = b2f(Vp[rb + row * 256 + col]);
          Et[col * KP3 + row] = f2b(c[rr] - v);
        }
      }
    }
    __syncthreads();
    // M += upd * K^T @ E    (tiles 16x16 over d x e, K padded to 96)
    for (int t = wv; t < 256; t += 16){
      int mt = t >> 4, nt = t & 15;
      int m = mt * 16 + ln, n = nt * 16 + ln;
      v8f c = {};
      #pragma unroll
      for (int kb = 0; kb < KP3; kb += 32){
        v16b a = frag_ld(Kct + m * KP3, kb, g);  // (K^T)[d][c]
        v16b b = frag_ld(Et + n * KP3, kb, g);   // E^T[e][c] == B[c][e]
        c = wmma_bf16(a, b, c);
      }
      #pragma unroll
      for (int rr = 0; rr < 8; ++rr){
        int row = mt * 16 + rr + 8 * g, col = nt * 16 + ln; // (d, e)
        int o = col * 256 + row;                            // Mt[e][d]
        Mt[o] = f2b(b2f(Mt[o]) + upd * c[rr]);
      }
    }
    __syncthreads();
  }
}

// =====================================================================
// 5) Merge streams back to y (B,H,W,512) + bf16 copy
// =====================================================================
__global__ void __launch_bounds__(256)
k_build_y(const float* __restrict__ xo0, const float* __restrict__ xo1,
          float* __restrict__ y, u16* __restrict__ yb){
  long t = blockIdx.x;
  int c = threadIdx.x;
  int b = (int)(t / NPIX);
  int r = (int)(t - (long)b * NPIX);
  int h = r >> 7, w = r & 127;
  long i0 = ((long)b * NTOK + (long)(h * NWIN + (w >> 6)) * AWL + (w & 63)) * 256 + c;
  long i1 = ((long)b * NTOK + (long)(w * NWIN + (h >> 6)) * AWL + (h & 63)) * 256 + c;
  float v0 = xo0[i0], v1 = xo1[i1];
  long o = t * DIMF;
  y[o + c] = v0;  y[o + c + 256] = v1;
  yb[o + c] = f2b(v0); yb[o + c + 256] = f2b(v1);
}

// =====================================================================
// 6) Fused FFN: y += gelu(y@W1)@W2, hidden in 128-chunks
// =====================================================================
__global__ void __launch_bounds__(256)
k_ffn(const u16* __restrict__ yb, float* __restrict__ y,
      const float* __restrict__ W1, const float* __restrict__ W2){
  __shared__ __align__(16) u16 Ys[32 * 512];
  __shared__ __align__(16) u16 Bt1[128 * 512];
  __shared__ __align__(16) u16 Hs[32 * 128];
  __shared__ __align__(16) u16 Bt2[512 * 128];
  int tid = threadIdx.x;
  int wv = tid >> 5, lane = tid & 31, g = lane >> 4, ln = lane & 15;
  long rbase = (long)blockIdx.x * 32;
  {
    const u16* src = yb + rbase * 512;
    for (int i = tid; i < 32 * 512 / 8; i += 256)
      cp16(Ys + i * 8, src + i * 8);
    cp_wait();
  }
  v8f acc[8];
  v8f zero = {};
  #pragma unroll
  for (int i = 0; i < 8; ++i) acc[i] = zero;
  for (int hc = 0; hc < 2048; hc += 128){
    __syncthreads();
    for (int i = tid; i < 128 * 512; i += 256){
      Bt1[i] = f2b(W1[(long)(i & 511) * 2048 + hc + (i >> 9)]);
      Bt2[i] = f2b(W2[(long)(hc + (i & 127)) * 512 + (i >> 7)]);
    }
    __syncthreads();
    for (int t = wv; t < 16; t += 8){        // H = gelu(Y @ W1chunk)
      int mt = t >> 3, nt = t & 7;
      int m = mt * 16 + ln, n = nt * 16 + ln;
      v8f c = {};
      #pragma unroll
      for (int kb = 0; kb < 512; kb += 32){
        v16b a = frag_ld(Ys + m * 512, kb, g);
        v16b b = frag_ld(Bt1 + n * 512, kb, g);
        c = wmma_bf16(a, b, c);
      }
      #pragma unroll
      for (int rr = 0; rr < 8; ++rr){
        int row = mt * 16 + rr + 8 * g, col = nt * 16 + ln;
        float xg = c[rr];
        float t3 = 0.7978845608028654f * (xg + 0.044715f * xg * xg * xg);
        Hs[row * 128 + col] = f2b(0.5f * xg * (1.f + tanhf(t3)));
      }
    }
    __syncthreads();
    #pragma unroll
    for (int i = 0; i < 8; ++i){             // acc += H @ W2chunk
      int t = wv * 8 + i;
      int mt = t >> 5, nt = t & 31;
      int m = mt * 16 + ln, n = nt * 16 + ln;
      #pragma unroll
      for (int kb = 0; kb < 128; kb += 32){
        v16b a = frag_ld(Hs + m * 128, kb, g);
        v16b b = frag_ld(Bt2 + n * 128, kb, g);
        acc[i] = wmma_bf16(a, b, acc[i]);
      }
    }
  }
  #pragma unroll
  for (int i = 0; i < 8; ++i){
    int t = wv * 8 + i;
    int mt = t >> 5, nt = t & 31;
    #pragma unroll
    for (int rr = 0; rr < 8; ++rr){
      int row = mt * 16 + rr + 8 * g, col = nt * 16 + ln;
      long o = (rbase + row) * 512 + col;
      y[o] += acc[i][rr];
    }
  }
}

// =====================================================================
// 7) RMSNorm -> bf16 (reuses yb buffer)
// =====================================================================
__global__ void __launch_bounds__(256)
k_rmsnorm(const float* __restrict__ y, const float* __restrict__ gamma,
          u16* __restrict__ ybn){
  __shared__ float red[256];
  long t = blockIdx.x;
  int c = threadIdx.x;
  float a = y[t * 512 + c], b = y[t * 512 + c + 256];
  red[c] = a * a + b * b;
  __syncthreads();
  for (int s = 128; s > 0; s >>= 1){
    if (c < s) red[c] += red[c + s];
    __syncthreads();
  }
  float inv = rsqrtf(red[0] / 512.f + 1e-6f);
  ybn[t * 512 + c]       = f2b(a * inv * gamma[c]);
  ybn[t * 512 + c + 256] = f2b(b * inv * gamma[c + 256]);
}

// =====================================================================
// 8) Output projection: out = yn @ prjW + prjb (f32)
// =====================================================================
__global__ void __launch_bounds__(256)
k_outproj(const u16* __restrict__ ybn, const float* __restrict__ Wp,
          const float* __restrict__ bp, float* __restrict__ out){
  __shared__ __align__(16) u16 As[64 * 512];
  __shared__ __align__(16) u16 Bt[128 * 512];
  int tid = threadIdx.x;
  int wv = tid >> 5, lane = tid & 31, g = lane >> 4, ln = lane & 15;
  long rbase = (long)blockIdx.x * 64;
  {
    const u16* src = ybn + rbase * 512;
    for (int i = tid; i < 64 * 512 / 8; i += 256)
      cp16(As + i * 8, src + i * 8);
    cp_wait();
  }
  for (int nc = 0; nc < 512; nc += 128){
    __syncthreads();
    for (int i = tid; i < 128 * 512; i += 256)
      Bt[i] = f2b(Wp[(long)(i & 511) * 512 + nc + (i >> 9)]);
    __syncthreads();
    for (int t = wv; t < 32; t += 8){
      int mt = t >> 3, nt = t & 7;
      int m = mt * 16 + ln, n = nt * 16 + ln;
      v8f c = {};
      #pragma unroll
      for (int kb = 0; kb < 512; kb += 32){
        v16b a = frag_ld(As + m * 512, kb, g);
        v16b b = frag_ld(Bt + n * 512, kb, g);
        c = wmma_bf16(a, b, c);
      }
      #pragma unroll
      for (int rr = 0; rr < 8; ++rr){
        int row = mt * 16 + rr + 8 * g, col = nc + nt * 16 + ln;
        out[(rbase + row) * 512 + col] = c[rr] + bp[col];
      }
    }
  }
}

// =====================================================================
extern "C" void kernel_launch(void* const* d_in, const int* in_sizes, int n_in,
                              void* d_out, int out_size, void* d_ws, size_t ws_size,
                              hipStream_t stream) {
  const float* x      = (const float*)d_in[0];
  const float* memsh0 = (const float*)d_in[4];
  const float* memsh1 = (const float*)d_in[5];
  const float* memsv0 = (const float*)d_in[6];
  const float* memsv1 = (const float*)d_in[7];
  const float* pe0    = (const float*)d_in[8];
  const float* pe1    = (const float*)d_in[9];
  const float* Wq0 = (const float*)d_in[10];
  const float* Wk0 = (const float*)d_in[11];
  const float* Wv0 = (const float*)d_in[12];
  const float* Wo0 = (const float*)d_in[13];
  const float* Wq1 = (const float*)d_in[14];
  const float* Wk1 = (const float*)d_in[15];
  const float* Wv1 = (const float*)d_in[16];
  const float* Wo1 = (const float*)d_in[17];
  const float* Mq0 = (const float*)d_in[18];
  const float* Mk0 = (const float*)d_in[19];
  const float* Mv0 = (const float*)d_in[20];
  const float* Mq1 = (const float*)d_in[21];
  const float* Mk1 = (const float*)d_in[22];
  const float* Mv1 = (const float*)d_in[23];
  const float* ffW1 = (const float*)d_in[24];
  const float* ffW2 = (const float*)d_in[25];
  const float* gamma = (const float*)d_in[26];
  const float* prjW  = (const float*)d_in[27];
  const float* prjb  = (const float*)d_in[28];

  char* ws = (char*)d_ws;
  size_t off = 0;
  auto alloc = [&](size_t bytes) -> void* {
    void* p = ws + off;
    off += (bytes + 255) & ~(size_t)255;
    return p;
  };
  const size_t STOK = (size_t)NTS * D2F;
  float* xo0  = (float*)alloc(STOK * 4);
  float* xo1  = (float*)alloc(STOK * 4);
  u16*  xmb0  = (u16*) alloc(STOK * 2);
  u16*  xmb1  = (u16*) alloc(STOK * 2);
  u16*  q0 = (u16*)alloc(STOK * 2);
  u16*  k0 = (u16*)alloc(STOK * 2);
  u16*  v0 = (u16*)alloc(STOK * 2);
  u16*  q1 = (u16*)alloc(STOK * 2);
  u16*  k1 = (u16*)alloc(STOK * 2);
  u16*  v1 = (u16*)alloc(STOK * 2);
  float* y  = (float*)alloc((size_t)TOKY * DIMF * 4);
  u16*  yb  = (u16*) alloc((size_t)TOKY * DIMF * 2);
  (void)in_sizes; (void)n_in; (void)out_size; (void)ws_size;

  k_build_xm<<<NTS, 256, 0, stream>>>(x, memsh0, memsh1, pe0, xo0, xmb0, 0);
  k_build_xm<<<NTS, 256, 0, stream>>>(x, memsv0, memsv1, pe1, xo1, xmb1, 1);
  k_wattn<<<512, 256, 0, stream>>>(xmb0, xo0, Wq0, Wk0, Wv0, Wo0);
  k_wattn<<<512, 256, 0, stream>>>(xmb1, xo1, Wq1, Wk1, Wv1, Wo1);
  k_proj256<<<NTS / 64, 256, 0, stream>>>(xmb0, Mq0, q0);
  k_proj256<<<NTS / 64, 256, 0, stream>>>(xmb0, Mk0, k0);
  k_proj256<<<NTS / 64, 256, 0, stream>>>(xmb0, Mv0, v0);
  k_proj256<<<NTS / 64, 256, 0, stream>>>(xmb1, Mq1, q1);
  k_proj256<<<NTS / 64, 256, 0, stream>>>(xmb1, Mk1, k1);
  k_proj256<<<NTS / 64, 256, 0, stream>>>(xmb1, Mv1, v1);
  k_nmem<<<2, 512, 0, stream>>>(q0, k0, v0, xo0);
  k_nmem<<<2, 512, 0, stream>>>(q1, k1, v1, xo1);
  k_build_y<<<TOKY, 256, 0, stream>>>(xo0, xo1, y, yb);
  k_ffn<<<TOKY / 32, 256, 0, stream>>>(yb, y, ffW1, ffW2);
  k_rmsnorm<<<TOKY, 256, 0, stream>>>(y, gamma, yb);
  k_outproj<<<TOKY / 64, 256, 0, stream>>>(yb, prjW, prjb, (float*)d_out);
}
